// MoE_75368086110256
// MI455X (gfx1250) — matrix operand order repaired and checked by reading, not compiled, not run on hardware
//
#include <hip/hip_runtime.h>
#include <math.h>
#include <stdint.h>

// ---------------------------------------------------------------------------
// MoE (top-2 of 8 experts + shared MLP) for gfx1250 / MI455X.
// GEMMs on v_wmma_f32_16x16x32_f16; x-tile staged to LDS with
// global_load_async_to_lds_b128 (ASYNCcnt) — CDNA5 async path.
// ---------------------------------------------------------------------------

typedef _Float16 v8h  __attribute__((ext_vector_type(8)));
typedef _Float16 v16h __attribute__((ext_vector_type(16)));
typedef float    v8f  __attribute__((ext_vector_type(8)));

#define NTOK  32768   // B*T*H*W = 2*4*64*64
#define D_DIM 512
#define I_DIM 2048
#define N_EXP 8
#define XS    520     // padded LDS stride (halves): 1040B = 260 dwords, 260%64=4

static __device__ inline v16h cat8(v8h a, v8h b) {
  return __builtin_shufflevector(a, b, 0,1,2,3,4,5,6,7,8,9,10,11,12,13,14,15);
}

static __device__ inline float gelu_exact(float v) {
  // jax.nn.gelu(approximate=False): 0.5*x*(1+erf(x/sqrt(2)))
  return 0.5f * v * (1.0f + erff(v * 0.70710678118654752440f));
}

// ---------------------------------------------------------------------------
// Gate: one wave32 per token. fp32 scores -> softmax -> top-2 -> renorm.
// Also converts x to f16 (each element touched exactly once).
// ---------------------------------------------------------------------------
__global__ __launch_bounds__(256)
void k_gate(const float* __restrict__ x, const float* __restrict__ gw,
            _Float16* __restrict__ xh, float* __restrict__ wfull)
{
  const int tid  = threadIdx.x;
  const int wave = tid >> 5;
  const int lane = tid & 31;
  const int t = blockIdx.x * 8 + wave;
  if (t >= NTOK) return;

  const float* xr = x + (size_t)t * D_DIM;
  float acc[N_EXP];
#pragma unroll
  for (int e = 0; e < N_EXP; ++e) acc[e] = 0.0f;

  for (int d = lane; d < D_DIM; d += 32) {
    const float xv = xr[d];
    xh[(size_t)t * D_DIM + d] = (_Float16)xv;
#pragma unroll
    for (int e = 0; e < N_EXP; ++e) acc[e] += xv * gw[e * D_DIM + d];
  }
#pragma unroll
  for (int e = 0; e < N_EXP; ++e) {
#pragma unroll
    for (int off = 16; off > 0; off >>= 1)
      acc[e] += __shfl_xor(acc[e], off, 32);
  }
  if (lane == 0) {
    // top-2 with lowest-index tie-break (matches jax.lax.top_k)
    int i1 = 0;
    for (int e = 1; e < N_EXP; ++e) if (acc[e] > acc[i1]) i1 = e;
    int i2 = (i1 == 0) ? 1 : 0;
    for (int e = 0; e < N_EXP; ++e)
      if (e != i1 && acc[e] > acc[i2]) i2 = e;
    // renormalized top-2 softmax weights (softmax monotone => same ranking)
    const float v2 = expf(acc[i2] - acc[i1]);
    const float s  = 1.0f + v2;
    const float w1 = 1.0f / s;
    const float w2 = v2 / s;
    for (int e = 0; e < N_EXP; ++e)
      wfull[(size_t)t * N_EXP + e] = (e == i1) ? w1 : ((e == i2) ? w2 : 0.0f);
  }
}

// ---------------------------------------------------------------------------
// fp32 -> f16 weight conversion
// ---------------------------------------------------------------------------
__global__ void k_cvt(const float* __restrict__ src, _Float16* __restrict__ dst, int n)
{
  for (int i = blockIdx.x * blockDim.x + threadIdx.x; i < n;
       i += gridDim.x * blockDim.x)
    dst[i] = (_Float16)src[i];
}

// ---------------------------------------------------------------------------
// Per-expert token lists, built in increasing token order (deterministic).
// One wave32 per expert.
// ---------------------------------------------------------------------------
__global__ void k_lists(const float* __restrict__ wfull,
                        int* __restrict__ lists, int* __restrict__ counts)
{
  const int e    = blockIdx.x;
  const int lane = threadIdx.x;   // 32 threads
  const unsigned long long ltmask = (1ull << lane) - 1ull;
  int base = 0;
  for (int c = 0; c < NTOK; c += 32) {
    const int t = c + lane;
    const bool f = wfull[(size_t)t * N_EXP + e] > 0.0f;
    const unsigned long long m = __ballot(f);
    if (f) lists[(size_t)e * NTOK + base + __popcll(m & ltmask)] = t;
    base += (int)__popcll(m);
  }
  if (lane == 0) counts[e] = base;
}

// ---------------------------------------------------------------------------
// Fused MLP tile kernel: out_tile[64 tok x 512] = fc2( gelu( fc1(x) ) )
//   list == nullptr : shared expert, out = result + b2      (writes all of out)
//   list != nullptr : routed expert, out += scale * (result + b2)
//
// Block: 256 threads = 8 waves; tile = 64 tokens x 512 output dims.
// The gathered 64x512 f16 x-tile is staged to LDS ONCE per block via
// global_load_async_to_lds_b128 (removes the 64x-redundant global A traffic),
// then per I-chunk of 64: each wave computes two fc1 h-tiles with WMMA
// (A from s_x, B = w1 rows from global), GELUs into s_h, and all waves
// accumulate fc2 (A from s_h, B = w2 rows from global), 128 acc VGPRs/wave.
//
// WMMA f16 fragment layouts (ISA 7.12.2):
//   A 16x32: g = lane>>4; row = lane&15; K run [g*16 .. +15] as loaded here (not ISA table order)
//   B 32x16: col = lane&15; K run [g*16 .. +15], same order as A
//   C 16x16: vgpr i, group g -> (M = i + 8*g, N = lane&15)
// ---------------------------------------------------------------------------
__global__ __launch_bounds__(256)
void k_mlp(const _Float16* __restrict__ xh,
           const _Float16* __restrict__ w1, const float* __restrict__ b1,
           const _Float16* __restrict__ w2, const float* __restrict__ b2,
           float* __restrict__ out,
           const int* __restrict__ list, const int* __restrict__ countp,
           const float* __restrict__ wcol)
{
  __shared__ int      s_tok[64];
  __shared__ float    s_scale[64];
  __shared__ _Float16 s_x[64][XS];   // gathered x tile (64 x 512, padded)
  __shared__ _Float16 s_h[64][80];   // 64x64 h-chunk, padded stride

  const int tid  = threadIdx.x;
  const int wave = tid >> 5;
  const int lane = tid & 31;
  const int g    = lane >> 4;
  const int ln   = lane & 15;

  int cnt = NTOK;
  const int mbase = blockIdx.x * 64;
  if (list) {
    cnt = *countp;
    if (mbase >= cnt) return;      // uniform early-exit, before any barrier
  }
  int nrows = cnt - mbase;
  if (nrows > 64) nrows = 64;

  if (tid < 64) {
    int t;
    if (list) t = (tid < nrows) ? list[mbase + tid] : list[mbase]; // pad: dup row0
    else      t = mbase + tid;
    s_tok[tid]   = t;
    s_scale[tid] = (list && tid < nrows) ? wcol[(size_t)t * N_EXP] : 0.0f;
  }
  __syncthreads();

  // ---- stage gathered x tile to LDS via async global->LDS (ASYNCcnt) ----
  // 64 rows x 1024B, one 16B transfer per op: 4096 ops / 256 threads.
  for (int c = tid; c < 64 * 64; c += 256) {
    const int row = c >> 6;
    const int ko  = (c & 63) * 8;                    // halves (16B granules)
    const _Float16* src = xh + (size_t)s_tok[row] * D_DIM + ko;
    const unsigned lds = (unsigned)(uintptr_t)&s_x[row][ko];
    asm volatile("global_load_async_to_lds_b128 %0, %1, off"
                 :: "v"(lds), "v"((unsigned long long)(uintptr_t)src)
                 : "memory");
  }
  asm volatile("s_wait_asynccnt 0" ::: "memory");
  __syncthreads();

  v8f oacc[4][4];
#pragma unroll
  for (int a = 0; a < 4; ++a)
#pragma unroll
    for (int b = 0; b < 4; ++b)
      oacc[a][b] = (v8f){};

  const int wbase = wave * 64;     // this wave's output-column base

  for (int ic = 0; ic < I_DIM; ic += 64) {
    // ---- phase A: h-chunk = gelu(x @ w1_chunk^T + b1) --------------------
#pragma unroll
    for (int u = 0; u < 2; ++u) {
      const int tt = wave * 2 + u;     // 16 tiles / 8 waves
      const int tm = tt >> 2;
      const int tn = tt & 3;
      const _Float16* ap = &s_x[tm * 16 + ln][0];
      const int icol = ic + tn * 16 + ln;          // I index = row of w1
      const _Float16* bp = w1 + (size_t)icol * D_DIM;
      v8f hacc = (v8f){};
#pragma unroll
      for (int k0 = 0; k0 < D_DIM; k0 += 32) {
        const v8h a0  = *(const v8h*)(ap + k0 + g * 16);
        const v8h a1  = *(const v8h*)(ap + k0 + g * 16 + 8);
        const v8h b0  = *(const v8h*)(bp + k0 + g * 16);
        const v8h b1v = *(const v8h*)(bp + k0 + g * 16 + 8);
        hacc = __builtin_amdgcn_wmma_f32_16x16x32_f16(
            false, cat8(a0, a1), false, cat8(b0, b1v),
            (short)0, hacc, false, false);
      }
      const float bias = b1[icol];
#pragma unroll
      for (int i = 0; i < 8; ++i) {
        const float v = gelu_exact(hacc[i] + bias);
        s_h[tm * 16 + i + g * 8][tn * 16 + ln] = (_Float16)v;
      }
    }
    __syncthreads();

    // ---- phase B: oacc += h-chunk @ w2_chunk^T ---------------------------
#pragma unroll
    for (int k0 = 0; k0 < 64; k0 += 32) {
      v16h Bf[4];
#pragma unroll
      for (int tn2 = 0; tn2 < 4; ++tn2) {
        const int col = wbase + tn2 * 16 + ln;     // D index = row of w2
        const _Float16* bp = w2 + (size_t)col * I_DIM + ic + k0 + g * 16;
        Bf[tn2] = cat8(*(const v8h*)bp, *(const v8h*)(bp + 8));
      }
#pragma unroll
      for (int tm = 0; tm < 4; ++tm) {
        const _Float16* hp = &s_h[tm * 16 + ln][0];
        const v16h A = cat8(*(const v8h*)(hp + k0 + g * 16),
                            *(const v8h*)(hp + k0 + g * 16 + 8));
#pragma unroll
        for (int tn2 = 0; tn2 < 4; ++tn2)
          oacc[tm][tn2] = __builtin_amdgcn_wmma_f32_16x16x32_f16(
              false, A, false, Bf[tn2], (short)0, oacc[tm][tn2], false, false);
      }
    }
    __syncthreads();
  }

  // ---- epilogue ----------------------------------------------------------
#pragma unroll
  for (int tn2 = 0; tn2 < 4; ++tn2) {
    const int col = wbase + tn2 * 16 + ln;
    const float bias = b2[col];
#pragma unroll
    for (int tm = 0; tm < 4; ++tm) {
#pragma unroll
      for (int i = 0; i < 8; ++i) {
        const int m = tm * 16 + i + g * 8;
        if (m < nrows) {
          const float v = oacc[tm][tn2][i] + bias;
          float* op = out + (size_t)s_tok[m] * D_DIM + col;
          if (list) *op += s_scale[m] * v;   // token unique per expert list
          else      *op = v;                 // shared pass initializes out
        }
      }
    }
  }
}

// ---------------------------------------------------------------------------
// Host launcher
// ---------------------------------------------------------------------------
extern "C" void kernel_launch(void* const* d_in, const int* in_sizes, int n_in,
                              void* d_out, int out_size, void* d_ws, size_t ws_size,
                              hipStream_t stream)
{
  (void)in_sizes; (void)n_in; (void)out_size; (void)ws_size;

  const float* x      = (const float*)d_in[0];
  const float* gate_w = (const float*)d_in[1];
  const float* fc1_w  = (const float*)d_in[2];
  const float* fc1_b  = (const float*)d_in[3];
  const float* fc2_w  = (const float*)d_in[4];
  const float* fc2_b  = (const float*)d_in[5];
  const float* sfc1_w = (const float*)d_in[6];
  const float* sfc1_b = (const float*)d_in[7];
  const float* sfc2_w = (const float*)d_in[8];
  const float* sfc2_b = (const float*)d_in[9];
  float* out = (float*)d_out;

  char* p = (char*)d_ws;
  _Float16* xh    = (_Float16*)p; p += (size_t)NTOK * D_DIM * 2;
  _Float16* fc1h  = (_Float16*)p; p += (size_t)N_EXP * I_DIM * D_DIM * 2;
  _Float16* fc2h  = (_Float16*)p; p += (size_t)N_EXP * D_DIM * I_DIM * 2;
  _Float16* sfc1h = (_Float16*)p; p += (size_t)I_DIM * D_DIM * 2;
  _Float16* sfc2h = (_Float16*)p; p += (size_t)D_DIM * I_DIM * 2;
  float* wfull    = (float*)p;    p += (size_t)NTOK * N_EXP * 4;
  int* lists      = (int*)p;      p += (size_t)N_EXP * NTOK * 4;
  int* counts     = (int*)p;      p += 256;

  // 1) gating + x->f16
  k_gate<<<NTOK / 8, 256, 0, stream>>>(x, gate_w, xh, wfull);
  // 2) weights -> f16
  k_cvt<<<2048, 256, 0, stream>>>(fc1_w,  fc1h,  N_EXP * I_DIM * D_DIM);
  k_cvt<<<2048, 256, 0, stream>>>(fc2_w,  fc2h,  N_EXP * D_DIM * I_DIM);
  k_cvt<<<1024, 256, 0, stream>>>(sfc1_w, sfc1h, I_DIM * D_DIM);
  k_cvt<<<1024, 256, 0, stream>>>(sfc2_w, sfc2h, D_DIM * I_DIM);
  // 3) deterministic per-expert token lists
  k_lists<<<N_EXP, 32, 0, stream>>>(wfull, lists, counts);
  // 4) shared expert (writes every element of out)
  k_mlp<<<NTOK / 64, 256, 0, stream>>>(xh, sfc1h, sfc1_b, sfc2h, sfc2_b, out,
                                       nullptr, nullptr, nullptr);
  // 5) routed experts, serialized on stream => deterministic accumulation
  for (int e = 0; e < N_EXP; ++e)
    k_mlp<<<NTOK / 64, 256, 0, stream>>>(
        xh,
        fc1h + (size_t)e * I_DIM * D_DIM, fc1_b + (size_t)e * I_DIM,
        fc2h + (size_t)e * D_DIM * I_DIM, fc2_b + (size_t)e * D_DIM,
        out, lists + (size_t)e * NTOK, counts + e, wfull + e);
}
